// CSS_MIL_25383256720176
// MI455X (gfx1250) — matrix operand order repaired and checked by reading, not compiled, not run on hardware
//
#include <hip/hip_runtime.h>
#include <stdint.h>

typedef unsigned short u16;
typedef __attribute__((ext_vector_type(16))) __bf16 v16bf;
typedef __attribute__((ext_vector_type(8)))  __bf16 v8bf;
typedef __attribute__((ext_vector_type(8)))  float  v8f;

#define L_SEQ   8200
#define LP      8208     // padded to multiple of 16
#define DM      512
#define DI      1024
#define DS      128
#define RNK     32
#define NX      288      // RNK + 2*DS (true x_proj width)
#define NXP     320      // padded to multiple of 64 for guard-free WMMA tiles
#define NCLS    8
#define CHUNK   1024
#define STRT    1025     // chunk + 1

// ---------- helpers ----------
__device__ __forceinline__ u16 f2bf(float f) {
  union { float f; uint32_t u; } v; v.f = f;
  uint32_t r = v.u + 0x7FFFu + ((v.u >> 16) & 1u);
  return (u16)(r >> 16);
}
__device__ __forceinline__ float bf2f(u16 h) {
  union { uint32_t u; float f; } v; v.u = ((uint32_t)h) << 16;
  return v.f;
}
__device__ __forceinline__ float sigm(float x){ return 1.f/(1.f+__expf(-x)); }
__device__ __forceinline__ float softp(float x){ return (x>15.f)?x:__logf(1.f+__expf(x)); }

// A fragment: 16-bit A 16x32 lane layout (two b128 loads)
__device__ __forceinline__ v16bf ld_a(const u16* __restrict__ arow, int half, int k0) {
  v8bf lo = *(const v8bf*)(arow + k0 + half * 16);
  v8bf hi = *(const v8bf*)(arow + k0 + half * 16 + 8);
  return __builtin_shufflevector(lo, hi, 0,1,2,3,4,5,6,7,8,9,10,11,12,13,14,15);
}
// B fragment: 16-bit B 32x16 lane layout from pre-transposed Bt (two b128 loads)
__device__ __forceinline__ v16bf ld_b(const u16* __restrict__ bp, int k0) {
  v8bf lo = *(const v8bf*)(bp + k0);
  v8bf hi = *(const v8bf*)(bp + k0 + 8);
  return __builtin_shufflevector(lo, hi, 0,1,2,3,4,5,6,7,8,9,10,11,12,13,14,15);
}
#define WMMA_BF16(a, b, c) \
  __builtin_amdgcn_wmma_f32_16x16x32_bf16(false, (a), false, (b), (short)0, (c), false, false)

// ---------- generic bf16 WMMA GEMM: C[MxN] = A[MxK] * Bt[NxK]^T ----------
// Requirements: M%16==0, N%64==0, K%32==0. Wave tile: 16 rows x 64 cols.
// Wave id forced uniform (readfirstlane): scalar branches only, EXEC all-ones
// around every WMMA. Steady-state loop is one straight-line basic block:
// load Y(k+32) | WMMA X | load X(k+64) | WMMA Y -- two ping-pong fragment
// sets refilled in place (no rotation copies), consistent pending-load counts
// on the back-edge so the waitcnt pass can use partial waits.
__global__ void __launch_bounds__(256)
k_gemm_wmma(const u16* __restrict__ A, const u16* __restrict__ Bt,
            float* __restrict__ Cf, u16* __restrict__ Cb,
            int M, int N, int K)
{
  const int wave = __builtin_amdgcn_readfirstlane((int)(threadIdx.x >> 5));
  const int lane = threadIdx.x & 31;
  const int half = lane >> 4;
  const int l15  = lane & 15;
  const int tiles_n = N >> 6;
  const int tiles_m = M >> 4;
  const int gw = blockIdx.x * 8 + wave;          // uniform (SGPR)
  if (gw >= tiles_m * tiles_n) return;           // scalar branch
  const int mt = gw / tiles_n;
  const int nt = gw % tiles_n;
  const int n0 = nt * 64;

  v8f acc0 = {}, acc1 = {}, acc2 = {}, acc3 = {};
  const u16* arow = A + (size_t)(mt * 16 + l15) * K;
  const u16* b0 = Bt + (size_t)(n0      + l15) * K + half * 16;
  const u16* b1 = Bt + (size_t)(n0 + 16 + l15) * K + half * 16;
  const u16* b2 = Bt + (size_t)(n0 + 32 + l15) * K + half * 16;
  const u16* b3 = Bt + (size_t)(n0 + 48 + l15) * K + half * 16;

  v16bf xa = ld_a(arow, half, 0);
  v16bf x0 = ld_b(b0, 0), x1 = ld_b(b1, 0), x2 = ld_b(b2, 0), x3 = ld_b(b3, 0);

  int k0 = 0;
  for (; k0 + 64 < K; k0 += 64) {
    // preload set Y @ k0+32 (always exists here)
    v16bf ya = ld_a(arow, half, k0 + 32);
    v16bf y0 = ld_b(b0, k0 + 32), y1 = ld_b(b1, k0 + 32);
    v16bf y2 = ld_b(b2, k0 + 32), y3 = ld_b(b3, k0 + 32);
    acc0 = WMMA_BF16(xa, x0, acc0);
    acc1 = WMMA_BF16(xa, x1, acc1);
    acc2 = WMMA_BF16(xa, x2, acc2);
    acc3 = WMMA_BF16(xa, x3, acc3);
    // refill set X @ k0+64 (always exists here; in place, no copies)
    xa = ld_a(arow, half, k0 + 64);
    x0 = ld_b(b0, k0 + 64); x1 = ld_b(b1, k0 + 64);
    x2 = ld_b(b2, k0 + 64); x3 = ld_b(b3, k0 + 64);
    acc0 = WMMA_BF16(ya, y0, acc0);
    acc1 = WMMA_BF16(ya, y1, acc1);
    acc2 = WMMA_BF16(ya, y2, acc2);
    acc3 = WMMA_BF16(ya, y3, acc3);
  }
  // tail: one or two 32-chunks remain
  if (k0 + 32 < K) {
    v16bf ya = ld_a(arow, half, k0 + 32);
    v16bf y0 = ld_b(b0, k0 + 32), y1 = ld_b(b1, k0 + 32);
    v16bf y2 = ld_b(b2, k0 + 32), y3 = ld_b(b3, k0 + 32);
    acc0 = WMMA_BF16(xa, x0, acc0);
    acc1 = WMMA_BF16(xa, x1, acc1);
    acc2 = WMMA_BF16(xa, x2, acc2);
    acc3 = WMMA_BF16(xa, x3, acc3);
    acc0 = WMMA_BF16(ya, y0, acc0);
    acc1 = WMMA_BF16(ya, y1, acc1);
    acc2 = WMMA_BF16(ya, y2, acc2);
    acc3 = WMMA_BF16(ya, y3, acc3);
  } else {
    acc0 = WMMA_BF16(xa, x0, acc0);
    acc1 = WMMA_BF16(xa, x1, acc1);
    acc2 = WMMA_BF16(xa, x2, acc2);
    acc3 = WMMA_BF16(xa, x3, acc3);
  }

#pragma unroll
  for (int r = 0; r < 8; ++r) {
    size_t rowb = (size_t)(mt * 16 + r + half * 8) * N + l15;
    if (Cf) {
      Cf[rowb + n0     ] = acc0[r];
      Cf[rowb + n0 + 16] = acc1[r];
      Cf[rowb + n0 + 32] = acc2[r];
      Cf[rowb + n0 + 48] = acc3[r];
    } else {
      Cb[rowb + n0     ] = f2bf(acc0[r]);
      Cb[rowb + n0 + 16] = f2bf(acc1[r]);
      Cb[rowb + n0 + 32] = f2bf(acc2[r]);
      Cb[rowb + n0 + 48] = f2bf(acc3[r]);
    }
  }
}

// ---------- fp32 -> bf16 elementwise ----------
__global__ void k_cvt_bf16(const float* __restrict__ s, u16* __restrict__ d, long n) {
  long i = (long)blockIdx.x * 256 + threadIdx.x;
  if (i < n) d[i] = f2bf(s[i]);
}

// ---------- weight transpose fp32[KxN(ld)] -> bf16[N_out x K], zero-pad n>=n_src ----------
__global__ void k_wt_bf16(const float* __restrict__ src, u16* __restrict__ dst,
                          int K, int n_out, int n_src, int src_ld, int col_off) {
  long idx = (long)blockIdx.x * 256 + threadIdx.x;
  if (idx >= (long)K * n_out) return;
  int n = (int)(idx / K), k = (int)(idx % K);
  u16 v = 0;
  if (n < n_src) v = f2bf(src[(size_t)k * src_ld + col_off + n]);
  dst[(size_t)n * K + k] = v;
}

// ---------- build interleaved seq (+bias) and its reverse, bf16, zero-pad ----------
__global__ void k_pack_seq(const float* __restrict__ h, const float* __restrict__ mb,
                           const float* __restrict__ tok,
                           u16* __restrict__ seq, u16* __restrict__ seqr) {
  long idx = (long)blockIdx.x * 256 + threadIdx.x;
  if (idx >= (long)LP * DM) return;
  int t = (int)(idx / DM), c = (int)(idx % DM);
  if (t >= L_SEQ) { seq[idx] = 0; seqr[idx] = 0; return; }
  int blk = t / STRT, r = t % STRT;
  float val;
  if (r == 0) val = tok[blk * DM + c];
  else        val = h[(size_t)(blk * CHUNK + (r - 1)) * DM + c] + mb[c];
  u16 bv = f2bf(val);
  seq[idx] = bv;
  seqr[(size_t)(L_SEQ - 1 - t) * DM + c] = bv;
}

// ---------- causal depthwise conv (width 4) + SiLU ----------
__global__ void k_conv_silu(const u16* __restrict__ xin, const float* __restrict__ cw,
                            const float* __restrict__ cb, u16* __restrict__ u) {
  long idx = (long)blockIdx.x * 256 + threadIdx.x;
  if (idx >= 2L * LP * DI) return;
  int d   = (int)(idx % DI);
  int t   = (int)((idx / DI) % LP);
  int dir = (int)(idx / ((long)DI * LP));
  if (t >= L_SEQ) { u[idx] = 0; return; }
  const u16* xp = xin + (size_t)dir * LP * DI;
  const float* w = cw + (size_t)(dir * DI + d) * 4;
  float acc = cb[dir * DI + d];
#pragma unroll
  for (int j = 0; j < 4; ++j) {
    int tt = t - 3 + j;
    if (tt >= 0) acc += w[j] * bf2f(xp[(size_t)tt * DI + d]);
  }
  acc = acc * sigm(acc);
  u[idx] = f2bf(acc);
}

// ---------- slice proj[:, :32] -> bf16 for dt GEMM ----------
__global__ void k_slice_dtin(const float* __restrict__ proj, u16* __restrict__ dtin) {
  long idx = (long)blockIdx.x * 256 + threadIdx.x;
  if (idx >= 2L * LP * RNK) return;
  int j   = (int)(idx % RNK);
  int t   = (int)((idx / RNK) % LP);
  int dir = (int)(idx / ((long)RNK * LP));
  dtin[idx] = f2bf(proj[(size_t)dir * LP * NXP + (size_t)t * NXP + j]);
}

// ---------- selective scan: 1 wave = 1 channel, 4 states/lane ----------
// Output C·h only at the 8 cls positions per direction.
__global__ void __launch_bounds__(256)
k_scan(const u16* __restrict__ dt_bf, const float* __restrict__ dt_b,
       const u16* __restrict__ u_bf, const float* __restrict__ proj,
       const float* __restrict__ A_log, float* __restrict__ ys) {
  int dir  = blockIdx.x >> 7;
  int grp  = blockIdx.x & 127;
  int wave = threadIdx.x >> 5;
  int lane = threadIdx.x & 31;
  int d    = grp * 8 + wave;

  const float* Al = A_log + ((size_t)dir * DI + d) * DS + lane * 4;
  float A0 = -__expf(Al[0]), A1 = -__expf(Al[1]);
  float A2 = -__expf(Al[2]), A3 = -__expf(Al[3]);
  float bias = dt_b[dir * DI + d];
  const u16* dtp = dt_bf + (size_t)dir * LP * DI + d;
  const u16* up  = u_bf  + (size_t)dir * LP * DI + d;
  const float* pj = proj + (size_t)dir * LP * NXP;

  float h0 = 0.f, h1 = 0.f, h2 = 0.f, h3 = 0.f;
  int nextT = dir ? (L_SEQ - 1 - STRT * 7) : 0;   // 1024 : 0
  int cls   = dir ? 7 : 0;

  for (int t = 0; t < L_SEQ; ++t) {
    float dtv = softp(bf2f(dtp[(size_t)t * DI]) + bias);
    float uv  = bf2f(up[(size_t)t * DI]);
    const float4 Bv = *(const float4*)(pj + (size_t)t * NXP + RNK + lane * 4);
    float dtu = dtv * uv;
    h0 = h0 * __expf(dtv * A0) + dtu * Bv.x;
    h1 = h1 * __expf(dtv * A1) + dtu * Bv.y;
    h2 = h2 * __expf(dtv * A2) + dtu * Bv.z;
    h3 = h3 * __expf(dtv * A3) + dtu * Bv.w;
    if (t == nextT) {
      const float4 Cv = *(const float4*)(pj + (size_t)t * NXP + RNK + DS + lane * 4);
      float p = h0 * Cv.x + h1 * Cv.y + h2 * Cv.z + h3 * Cv.w;
#pragma unroll
      for (int off = 16; off >= 1; off >>= 1) p += __shfl_xor(p, off, 32);
      if (lane == 0) ys[((size_t)dir * NCLS + cls) * DI + d] = p;
      nextT += STRT;
      cls += dir ? -1 : 1;
    }
  }
}

// ---------- z at cls rows: cls rows of seq are exactly the cls tokens ----------
__global__ void k_zcls(const float* __restrict__ tok, const float* __restrict__ inW,
                       float* __restrict__ zc) {
  int idx = blockIdx.x * 256 + threadIdx.x;   // 2*8*1024
  if (idx >= 2 * NCLS * DI) return;
  int d   = idx % DI;
  int i   = (idx / DI) % NCLS;
  int dir = idx / (DI * NCLS);
  const float* tk = tok + i * DM;
  const float* w  = inW + (size_t)dir * DM * 2 * DI + (DI + d);
  float acc = 0.f;
  for (int k = 0; k < DM; ++k) acc += tk[k] * w[(size_t)k * 2 * DI];
  zc[idx] = acc;
}

// ---------- y_cls = (ys + u*Dp) * silu(z), packed 16 rows (8 real + 8 zero) ----------
__global__ void k_ycls(const float* __restrict__ ys, const u16* __restrict__ u_bf,
                       const float* __restrict__ Dp, const float* __restrict__ zc,
                       u16* __restrict__ yc) {
  int idx = blockIdx.x * 256 + threadIdx.x;   // 2*16*1024
  if (idx >= 2 * 16 * DI) return;
  int d   = idx % DI;
  int i   = (idx / DI) % 16;
  int dir = idx / (16 * DI);
  if (i >= NCLS) { yc[idx] = 0; return; }
  int tpos = dir ? (L_SEQ - 1 - STRT * i) : STRT * i;
  float uv = bf2f(u_bf[((size_t)dir * LP + tpos) * DI + d]);
  float y  = ys[((size_t)dir * NCLS + i) * DI + d] + uv * Dp[dir * DI + d];
  float z  = zc[((size_t)dir * NCLS + i) * DI + d];
  y *= z * sigm(z);
  yc[idx] = f2bf(y);
}

// ---------- classifier layer 1: hid = relu(flat(8192) @ W(8192x512) + b) ----------
__global__ void k_cls1(const float* __restrict__ oc, const float* __restrict__ W,
                       const float* __restrict__ b, float* __restrict__ hid) {
  int j = blockIdx.x * 256 + threadIdx.x;
  if (j >= 512) return;
  float acc = b[j];
  for (int k = 0; k < 8192; ++k) {
    int i = k >> 10, cc = k & 1023;
    int dirk = cc >> 9, col = cc & 511;
    float v = oc[((size_t)dirk * 16 + i) * DM + col];
    acc += v * W[(size_t)k * 512 + j];
  }
  hid[j] = fmaxf(acc, 0.f);
}

// ---------- classifier layer 2 ----------
__global__ void k_cls2(const float* __restrict__ hid, const float* __restrict__ W,
                       const float* __restrict__ b, float* __restrict__ out) {
  int c = threadIdx.x;
  if (c >= 2) return;
  float acc = b[c];
  for (int j = 0; j < 512; ++j) acc += hid[j] * W[j * 2 + c];
  out[c] = acc;
}

extern "C" void kernel_launch(void* const* d_in, const int* in_sizes, int n_in,
                              void* d_out, int out_size, void* d_ws, size_t ws_size,
                              hipStream_t stream) {
  (void)in_sizes; (void)n_in; (void)out_size; (void)ws_size;
  const float* x        = (const float*)d_in[0];
  const float* map_W    = (const float*)d_in[1];
  const float* map_b    = (const float*)d_in[2];
  const float* cls_tok  = (const float*)d_in[3];
  const float* in_W     = (const float*)d_in[4];
  const float* conv_W   = (const float*)d_in[5];
  const float* conv_b   = (const float*)d_in[6];
  const float* xproj_W  = (const float*)d_in[7];
  const float* dtproj_W = (const float*)d_in[8];
  const float* dtproj_b = (const float*)d_in[9];
  const float* A_log    = (const float*)d_in[10];
  const float* Dp       = (const float*)d_in[11];
  const float* outp_W   = (const float*)d_in[12];
  const float* c1W      = (const float*)d_in[13];
  const float* c1b      = (const float*)d_in[14];
  const float* c2W      = (const float*)d_in[15];
  const float* c2b      = (const float*)d_in[16];

  char* ws = (char*)d_ws;
  size_t off = 0;
  auto alloc = [&](size_t bytes) -> char* {
    char* p = ws + off; off += (bytes + 255) & ~(size_t)255; return p;
  };
  auto cdiv = [](long a, long b) -> int { return (int)((a + b - 1) / b); };

  u16*   x_bf    = (u16*)  alloc(8192ull * 1024 * 2);
  u16*   mapWt   = (u16*)  alloc((size_t)DM * 1024 * 2);
  float* h_f     = (float*)alloc(8192ull * DM * 4);
  u16*   seq_bf  = (u16*)  alloc((size_t)LP * DM * 2);
  u16*   seqr_bf = (u16*)  alloc((size_t)LP * DM * 2);
  u16*   inxWt   = (u16*)  alloc(2ull * DI * DM * 2);
  u16*   xin_bf  = (u16*)  alloc(2ull * LP * DI * 2);
  u16*   u_bf    = (u16*)  alloc(2ull * LP * DI * 2);
  u16*   xpWt    = (u16*)  alloc(2ull * NXP * DI * 2);
  float* proj    = (float*)alloc(2ull * LP * NXP * 4);
  u16*   dtin_bf = (u16*)  alloc(2ull * LP * RNK * 2);
  u16*   dtpWt   = (u16*)  alloc(2ull * DI * RNK * 2);
  u16*   dt_bf   = (u16*)  alloc(2ull * LP * DI * 2);
  u16*   opWt    = (u16*)  alloc(2ull * DM * DI * 2);
  float* ys_cls  = (float*)alloc(2ull * NCLS * DI * 4);
  float* z_cls   = (float*)alloc(2ull * NCLS * DI * 4);
  u16*   ycls_bf = (u16*)  alloc(2ull * 16 * DI * 2);
  float* out_cls = (float*)alloc(2ull * 16 * DM * 4);
  float* hid     = (float*)alloc(512 * 4);

  auto gemm_grid = [](int M, int N) -> int {
    long t = (long)(M / 16) * (N / 64);
    return (int)((t + 7) / 8);
  };

  // 1) x -> bf16 ; map_W^T
  k_cvt_bf16<<<cdiv(8192L * 1024, 256), 256, 0, stream>>>(x, x_bf, 8192L * 1024);
  k_wt_bf16<<<cdiv(1024L * DM, 256), 256, 0, stream>>>(map_W, mapWt, 1024, DM, DM, DM, 0);
  // 2) embed GEMM: h = x @ map_W
  k_gemm_wmma<<<gemm_grid(8192, DM), 256, 0, stream>>>(x_bf, mapWt, h_f, nullptr, 8192, DM, 1024);
  // 3) build interleaved sequence (+bias, +cls tokens) forward & reversed
  k_pack_seq<<<cdiv((long)LP * DM, 256), 256, 0, stream>>>(h_f, map_b, cls_tok, seq_bf, seqr_bf);

  for (int dir = 0; dir < 2; ++dir) {
    // in_proj (x-half only): xin = seq_dir @ W[:, :1024]
    k_wt_bf16<<<cdiv(512L * DI, 256), 256, 0, stream>>>(
        in_W + (size_t)dir * 512 * 2048, inxWt + (size_t)dir * DI * DM, DM, DI, DI, 2048, 0);
    k_gemm_wmma<<<gemm_grid(LP, DI), 256, 0, stream>>>(
        dir ? seqr_bf : seq_bf, inxWt + (size_t)dir * DI * DM,
        nullptr, xin_bf + (size_t)dir * LP * DI, LP, DI, DM);
  }
  // 4) depthwise causal conv + silu (both dirs)
  k_conv_silu<<<cdiv(2L * LP * DI, 256), 256, 0, stream>>>(xin_bf, conv_W, conv_b, u_bf);

  for (int dir = 0; dir < 2; ++dir) {
    // x_proj: proj = u @ x_proj_W, N padded 288 -> 320 (pad weight rows are zero)
    k_wt_bf16<<<cdiv((long)NXP * DI, 256), 256, 0, stream>>>(
        xproj_W + (size_t)dir * 1024 * NX, xpWt + (size_t)dir * NXP * DI, DI, NXP, NX, NX, 0);
    k_gemm_wmma<<<gemm_grid(LP, NXP), 256, 0, stream>>>(
        u_bf + (size_t)dir * LP * DI, xpWt + (size_t)dir * NXP * DI,
        proj + (size_t)dir * LP * NXP, nullptr, LP, NXP, DI);
  }
  // 5) dt GEMM: dt_raw = proj[:, :32] @ dt_proj_W
  k_slice_dtin<<<cdiv(2L * LP * RNK, 256), 256, 0, stream>>>(proj, dtin_bf);
  for (int dir = 0; dir < 2; ++dir) {
    k_wt_bf16<<<cdiv(32L * DI, 256), 256, 0, stream>>>(
        dtproj_W + (size_t)dir * RNK * DI, dtpWt + (size_t)dir * DI * RNK, RNK, DI, DI, DI, 0);
    k_gemm_wmma<<<gemm_grid(LP, DI), 256, 0, stream>>>(
        dtin_bf + (size_t)dir * LP * RNK, dtpWt + (size_t)dir * DI * RNK,
        nullptr, dt_bf + (size_t)dir * LP * DI, LP, DI, RNK);
  }
  // 6) selective scan (softplus+bias fused), outputs only at 8 cls positions/dir
  k_scan<<<256, 256, 0, stream>>>(dt_bf, dtproj_b, u_bf, proj, A_log, ys_cls);
  // 7) gate z at cls rows (cls rows of seq == cls tokens)
  k_zcls<<<cdiv(2L * NCLS * DI, 256), 256, 0, stream>>>(cls_tok, in_W, z_cls);
  // 8) y_cls = (ys + u*Dp) * silu(z), packed 16x1024 (rows 8..15 zero)
  k_ycls<<<cdiv(2L * 16 * DI, 256), 256, 0, stream>>>(ys_cls, u_bf, Dp, z_cls, ycls_bf);
  // 9) out_proj at cls rows only: [16x1024] @ [1024x512]
  for (int dir = 0; dir < 2; ++dir) {
    k_wt_bf16<<<cdiv(1024L * DM, 256), 256, 0, stream>>>(
        outp_W + (size_t)dir * DI * DM, opWt + (size_t)dir * DM * DI, DI, DM, DM, DM, 0);
    k_gemm_wmma<<<gemm_grid(16, DM), 256, 0, stream>>>(
        ycls_bf + (size_t)dir * 16 * DI, opWt + (size_t)dir * DM * DI,
        out_cls + (size_t)dir * 16 * DM, nullptr, 16, DM, DI);
  }
  // 10) classifier
  k_cls1<<<2, 256, 0, stream>>>(out_cls, c1W, c1b, hid);
  k_cls2<<<1, 32, 0, stream>>>(hid, c2W, c2b, (float*)d_out);
}
